// Attention_45397804319204
// MI455X (gfx1250) — matrix operand order repaired and checked by reading, not compiled, not run on hardware
//
#include <hip/hip_runtime.h>
#include <hip/hip_bf16.h>

// ---------------------------------------------------------------------------
// Causal multi-head self-attention (GPT block) for gfx1250 / MI455X.
// fp16 WMMA (v_wmma_f32_16x16x32_f16) with f32 accumulation.
// All tile staging through the Tensor Data Mover (tensor_load_to_lds +
// s_wait_tensorcnt) into double-buffered, pad-interleaved (bank-conflict-free)
// LDS. Flash attention uses the transposed-S formulation so softmax rows map
// to lanes (wave32): per-lane max/sum + one xor-16 swizzle.
// ---------------------------------------------------------------------------

#define SEQ_T   4096
#define EMB_C   768
#define NHEAD   12
#define HSIZE   64
#define BATCH   2
#define QKV_N   (3 * EMB_C)      // 2304
#define MTOT    (BATCH * SEQ_T)  // 8192
#define KCHUNK  64
#define NCHUNK  (EMB_C / KCHUNK) // 12 (even)
#define SCALE_F 0.125f           // 1/sqrt(64)

// TDM pad codes: interval code k -> 2^(k+1) DWORDs, amount code a -> a+1 DWORDs
#define PAD_I32 4                // pad every 32 DWORDs (64 f16 row)
#define PAD_I16 3                // pad every 16 DWORDs (32 f16 row)
#define PAD_A4  3                // insert 4 DWORDs (8 f16)
#define BPITCH  72               // 64 f16 row + 8 f16 pad  (36 DW: conflict-free)
#define VPITCH  40               // 32 f16 row + 8 f16 pad  (20 DW: conflict-free)

typedef __attribute__((ext_vector_type(16))) _Float16 v16h;
typedef __attribute__((ext_vector_type(8)))  _Float16 v8h;
typedef __attribute__((ext_vector_type(8)))  float    v8f;
typedef __attribute__((ext_vector_type(4)))  unsigned v4u;
typedef __attribute__((ext_vector_type(8)))  int      v8i;
typedef __attribute__((ext_vector_type(4)))  int      v4i;

// Select LDS staging path at compile time (all probe-safe fallbacks).
#if __has_builtin(__builtin_amdgcn_tensor_load_to_lds)
#define STAGE_TDM 1
#elif __has_builtin(__builtin_amdgcn_global_load_async_to_lds_b128)
#define STAGE_ASYNC 1
#endif

// ----- WMMA fragment loaders (wave32; contiguous k = 16*(lane/16) + s in both A and B) ----------

// A-fragment 16x32 (MxK) f16 from row-major [rows][ld] (global or LDS).
__device__ inline v16h load_frag_a(const _Float16* base,
                                   int row0, int k0, int ld, int lane) {
    const int l  = lane & 15;
    const int hi = (lane >> 4) * 16;
    const _Float16* p = base + (size_t)(row0 + l) * (size_t)ld + k0 + hi;
    v8h c0 = *(const v8h*)(p);
    v8h c1 = *(const v8h*)(p + 8);
    v16h r;
#pragma unroll
    for (int i = 0; i < 8; ++i) { r[i] = c0[i]; r[8 + i] = c1[i]; }
    return r;
}

// B-fragment 32x16 (KxN) f16: output column n is row (col0+n) of [N][K] matrix.
__device__ inline v16h load_frag_b(const _Float16* base,
                                   int col0, int k0, int ld, int lane) {
    const int n  = lane & 15;
    const int hi = (lane >> 4) * 16;
    const _Float16* p = base + (size_t)(col0 + n) * (size_t)ld + k0 + hi;
    v8h c0 = *(const v8h*)(p);
    v8h c1 = *(const v8h*)(p + 8);
    v16h r;
#pragma unroll
    for (int i = 0; i < 8; ++i) { r[i] = c0[i]; r[8 + i] = c1[i]; }
    return r;
}

// B-fragment from an LDS-staged [64 rows][BPITCH] chunk.
__device__ inline v16h lds_frag_b(const _Float16* smem, int tsub, int k0loc,
                                  int lane) {
    const int n  = lane & 15;
    const int hi = (lane >> 4) * 16;
    const _Float16* p = smem + (tsub * 16 + n) * BPITCH + k0loc + hi;
    v8h c0 = *(const v8h*)(p);
    v8h c1 = *(const v8h*)(p + 8);
    v16h r;
#pragma unroll
    for (int i = 0; i < 8; ++i) { r[i] = c0[i]; r[8 + i] = c1[i]; }
    return r;
}

__device__ inline v8f wmma_16x16x32(v16h a, v16h b, v8f c) {
    return __builtin_amdgcn_wmma_f32_16x16x32_f16(
        false, a, false, b, (short)0, c, false, false);
}

// ----- Tensor Data Mover: generic padded 2-D tile stage --------------------

#if defined(STAGE_TDM)
// D# per CDNA5 ISA 8.3/8.4: tensor f16 [td1][stride0] (row len td0),
// tile [tile1 rows][tile0 cols] starting at src -> LDS dst with pad codes.
__device__ inline void tdm_stage2d(const _Float16* src, unsigned td0,
                                   unsigned td1, unsigned stride0,
                                   unsigned tile0, unsigned tile1,
                                   unsigned ivl, unsigned amt,
                                   _Float16* dst) {
    const unsigned long long ga = (unsigned long long)(size_t)src;
    const unsigned lds = (unsigned)(size_t)dst;   // flat LDS addr[31:0] = offset
    const unsigned long long g0lo = 1ull | ((unsigned long long)lds << 32);
    const unsigned long long g0hi =
        (ga & 0x01FFFFFFFFFFFFFFull) | (2ull << 62);
    const unsigned long long d0 =
        (1ull << 16) | (1ull << 20) |
        ((unsigned long long)ivl << 22) | ((unsigned long long)amt << 25) |
        ((unsigned long long)(td0 & 0xFFFF) << 48);
    const unsigned long long d1 =
        ((unsigned long long)(td0 >> 16) & 0xFFFF) |
        ((unsigned long long)td1 << 16) | ((unsigned long long)tile0 << 48);
    const unsigned long long d2 =
        (unsigned long long)tile1 | ((unsigned long long)stride0 << 32);
    const unsigned long long d3 = 0;
    v4u g0 = {(unsigned)g0lo, (unsigned)(g0lo >> 32),
              (unsigned)g0hi, (unsigned)(g0hi >> 32)};
    v8i g1 = {(int)d0, (int)(d0 >> 32), (int)d1, (int)(d1 >> 32),
              (int)d2, (int)(d2 >> 32), (int)d3, (int)(d3 >> 32)};
    v4i z4 = {0, 0, 0, 0};
    v8i z8 = {0, 0, 0, 0, 0, 0, 0, 0};
    __builtin_amdgcn_tensor_load_to_lds(g0, g1, z4, z4, z8, 0);
}
#endif

#if defined(STAGE_ASYNC)
__device__ inline void async_cp16(const _Float16* g, _Float16* l) {
    __builtin_amdgcn_global_load_async_to_lds_b128(
        (const __attribute__((address_space(1))) void*)g,
        (__attribute__((address_space(3))) void*)l, 0, 0);
}
#endif

__device__ inline void stage_wait() {
#if defined(STAGE_TDM)
    __builtin_amdgcn_s_wait_tensorcnt(0);
#elif defined(STAGE_ASYNC)
#if __has_builtin(__builtin_amdgcn_s_wait_asynccnt)
    __builtin_amdgcn_s_wait_asynccnt(0);
#else
    asm volatile("s_wait_asynccnt 0x0" ::: "memory");
#endif
#endif
}

// ----- GEMM weight-chunk staging: 64 rows x KCHUNK f16 ---------------------

__device__ inline void gemm_stage(const _Float16* __restrict__ W, int n0,
                                  int k0, _Float16* dst, int wrows,
                                  int tid, int wave) {
#if defined(STAGE_TDM)
    if (wave == 0)
        tdm_stage2d(W + (size_t)n0 * EMB_C + k0, EMB_C, (unsigned)wrows,
                    EMB_C, KCHUNK, 64, PAD_I32, PAD_A4, dst);
    (void)tid;
#else
#pragma unroll
    for (int i = 0; i < 2; ++i) {                 // 512 x 16B = 8KB chunk
        const int idx = tid + i * 256;
        const int row = idx >> 3, ch = idx & 7;
        const _Float16* g = W + (size_t)(n0 + row) * EMB_C + k0 + ch * 8;
        _Float16* l = dst + row * BPITCH + ch * 8;
#if defined(STAGE_ASYNC)
        async_cp16(g, l);
#else
        *(v8h*)l = *(const v8h*)g;
#endif
    }
    (void)wrows; (void)wave;
#endif
}

// One staged chunk: 2 K-steps x (2 A-frags x 4 B-frags) = 16 WMMA.
__device__ inline void gemm_chunk(const _Float16* __restrict__ A, int m0,
                                  int kbase, const _Float16* sbuf,
                                  v8f acc[2][4], int lane) {
#pragma unroll
    for (int s = 0; s < 2; ++s) {
        const int k0 = kbase + s * 32;
        v16h a0 = load_frag_a(A, m0,      k0, EMB_C, lane);
        v16h a1 = load_frag_a(A, m0 + 16, k0, EMB_C, lane);
        v16h b0 = lds_frag_b(sbuf, 0, s * 32, lane);
        v16h b1 = lds_frag_b(sbuf, 1, s * 32, lane);
        v16h b2 = lds_frag_b(sbuf, 2, s * 32, lane);
        v16h b3 = lds_frag_b(sbuf, 3, s * 32, lane);
        acc[0][0] = wmma_16x16x32(a0, b0, acc[0][0]);
        acc[1][0] = wmma_16x16x32(a1, b0, acc[1][0]);
        acc[0][1] = wmma_16x16x32(a0, b1, acc[0][1]);
        acc[1][1] = wmma_16x16x32(a1, b1, acc[1][1]);
        acc[0][2] = wmma_16x16x32(a0, b2, acc[0][2]);
        acc[1][2] = wmma_16x16x32(a1, b2, acc[1][2]);
        acc[0][3] = wmma_16x16x32(a0, b3, acc[0][3]);
        acc[1][3] = wmma_16x16x32(a1, b3, acc[1][3]);
    }
}

// Double-buffered mainloop: acc[2][4] += A[m0:+32, :] * W[n0:+64, :]^T
__device__ inline void gemm_mainloop(const _Float16* __restrict__ A,
                                     const _Float16* __restrict__ W,
                                     int m0, int n0, int wrows,
                                     _Float16 (*sB)[64 * BPITCH], v8f acc[2][4],
                                     int tid, int lane, int wave) {
    gemm_stage(W, n0, 0, sB[0], wrows, tid, wave);
    stage_wait();
    __syncthreads();
#pragma unroll 1
    for (int c = 0; c < NCHUNK; c += 2) {
        gemm_stage(W, n0, (c + 1) * KCHUNK, sB[1], wrows, tid, wave);
        gemm_chunk(A, m0, c * KCHUNK, sB[0], acc, lane);
        stage_wait();
        __syncthreads();
        if (c + 2 < NCHUNK)
            gemm_stage(W, n0, (c + 2) * KCHUNK, sB[0], wrows, tid, wave);
        gemm_chunk(A, m0, (c + 1) * KCHUNK, sB[1], acc, lane);
        stage_wait();
        __syncthreads();
    }
}

// ----- fp32 -> fp16 conversion --------------------------------------------

__global__ void cvt_f32_to_f16(const float* __restrict__ src,
                               _Float16* __restrict__ dst, int n) {
    int i = blockIdx.x * blockDim.x + threadIdx.x;
    int stride = gridDim.x * blockDim.x;
    for (; i < n; i += stride) dst[i] = (_Float16)src[i];
}

// ----- Kernel 2: QKV = x @ w_attn^T + b_attn, scatter to Q/K/V^T -----------
// 8 waves, block tile 256(M) x 64(N); wave tile 32 x 64.

__global__ __launch_bounds__(256)
void qkv_gemm(const _Float16* __restrict__ xh,
              const _Float16* __restrict__ wah,
              const float*    __restrict__ b_attn,
              _Float16* __restrict__ Qh,   // [B,H,T,HS], pre-scaled by SCALE
              _Float16* __restrict__ Kh,   // [B,H,T,HS]
              _Float16* __restrict__ Vt) { // [B,H,HS,T] (transposed)
    __shared__ __align__(16) _Float16 sB[2][64 * BPITCH];
    const int tid  = threadIdx.x;
    const int lane = tid & 31;
    const int wave = tid >> 5;
    const int m0 = blockIdx.x * 256 + wave * 32;
    const int n0 = blockIdx.y * 64;

    const v8f z = {0.f,0.f,0.f,0.f,0.f,0.f,0.f,0.f};
    v8f acc[2][4];
#pragma unroll
    for (int mh = 0; mh < 2; ++mh)
#pragma unroll
        for (int t = 0; t < 4; ++t) acc[mh][t] = z;

    gemm_mainloop(xh, wah, m0, n0, QKV_N, sB, acc, tid, lane, wave);

    const int hi8 = (lane >> 4) * 8;
    const int nc  = lane & 15;
#pragma unroll
    for (int mh = 0; mh < 2; ++mh) {
        const int r0  = m0 + mh * 16 + hi8;
        const int b   = r0 >> 12;             // 8-row pack never crosses batch
        const int tt0 = r0 & (SEQ_T - 1);
#pragma unroll
        for (int t = 0; t < 4; ++t) {
            const int d = n0 + t * 16 + nc;   // 0..2303
            const float bias = b_attn[d];
            if (d < EMB_C) {
                const int h = d >> 6, dp = d & 63;
                _Float16* q =
                    Qh + ((size_t)(b * NHEAD + h) * SEQ_T + tt0) * HSIZE + dp;
#pragma unroll
                for (int v = 0; v < 8; ++v)
                    q[(size_t)v * HSIZE] =
                        (_Float16)((acc[mh][t][v] + bias) * SCALE_F);
            } else if (d < 2 * EMB_C) {
                const int d2 = d - EMB_C;
                const int h = d2 >> 6, dp = d2 & 63;
                _Float16* k =
                    Kh + ((size_t)(b * NHEAD + h) * SEQ_T + tt0) * HSIZE + dp;
#pragma unroll
                for (int v = 0; v < 8; ++v)
                    k[(size_t)v * HSIZE] = (_Float16)(acc[mh][t][v] + bias);
            } else {
                const int d2 = d - 2 * EMB_C;
                const int h = d2 >> 6, dp = d2 & 63;
                v8h pk;                        // 8 consecutive t's: packed store
#pragma unroll
                for (int v = 0; v < 8; ++v)
                    pk[v] = (_Float16)(acc[mh][t][v] + bias);
                *(v8h*)(Vt + ((size_t)(b * NHEAD + h) * HSIZE + dp) * SEQ_T +
                        tt0) = pk;
            }
        }
    }
}

// ----- Kernel 3: flash attention with TDM-staged K/V tiles -----------------
// 4 waves, 64 queries per block. Per 32-key block (from LDS):
//   S^T(32x16)=K x Q^T (4 WMMA), online softmax, O^T += V^T x P^T (4 WMMA).

__device__ inline void flash_stage(const _Float16* __restrict__ Kb,
                                   const _Float16* __restrict__ Vb,
                                   int j0, _Float16* sK, _Float16* sV,
                                   int tid, int wave) {
#if defined(STAGE_TDM)
    if (wave == 0)   // K tile: 32 rows x 64, contiguous rows
        tdm_stage2d(Kb + (size_t)j0 * HSIZE, HSIZE, SEQ_T, HSIZE,
                    HSIZE, 32, PAD_I32, PAD_A4, sK);
    if (wave == 1)   // V^T tile: 64 rows x 32, row stride SEQ_T
        tdm_stage2d(Vb + j0, SEQ_T, HSIZE, SEQ_T,
                    32, HSIZE, PAD_I16, PAD_A4, sV);
    (void)tid;
#else
#pragma unroll
    for (int i = 0; i < 2; ++i) {                 // K: 32 rows x 8 x 16B
        const int idx = tid + i * 128;
        const int row = idx >> 3, ch = idx & 7;
        const _Float16* g = Kb + (size_t)(j0 + row) * HSIZE + ch * 8;
        _Float16* l = sK + row * BPITCH + ch * 8;
#if defined(STAGE_ASYNC)
        async_cp16(g, l);
#else
        *(v8h*)l = *(const v8h*)g;
#endif
    }
#pragma unroll
    for (int i = 0; i < 2; ++i) {                 // V: 64 rows x 4 x 16B
        const int idx = tid + i * 128;
        const int row = idx >> 2, ch = idx & 3;
        const _Float16* g = Vb + (size_t)row * SEQ_T + j0 + ch * 8;
        _Float16* l = sV + row * VPITCH + ch * 8;
#if defined(STAGE_ASYNC)
        async_cp16(g, l);
#else
        *(v8h*)l = *(const v8h*)g;
#endif
    }
    (void)wave;
#endif
}

__global__ __launch_bounds__(128)
void flash_attn(const _Float16* __restrict__ Qh,
                const _Float16* __restrict__ Kh,
                const _Float16* __restrict__ Vt,
                _Float16* __restrict__ AO) {   // [B,T,C] f16
    __shared__ __align__(16) _Float16 sK[2][32 * BPITCH];
    __shared__ __align__(16) _Float16 sV[2][HSIZE * VPITCH];
    const int tid  = threadIdx.x;
    const int lane = tid & 31;
    const int wave = tid >> 5;
    const int bh = blockIdx.y;
    const size_t base = (size_t)bh * SEQ_T * HSIZE;
    const _Float16* Qb = Qh + base;
    const _Float16* Kb = Kh + base;
    const _Float16* Vb = Vt + base;            // [HS][T]

    const int q0 = blockIdx.x * 64 + wave * 16;
    const int qcol = q0 + (lane & 15);
    const int hi8 = (lane >> 4) * 8;
    const bool isHi = (lane >= 16);

    const v16h bq_lo = load_frag_b(Qb, q0, 0,  HSIZE, lane);
    const v16h bq_hi = load_frag_b(Qb, q0, 32, HSIZE, lane);

    const float NEG_INF = -__builtin_inff();
    float m_i = NEG_INF;
    float l_i = 0.f;
    const v8f z = {0.f,0.f,0.f,0.f,0.f,0.f,0.f,0.f};
    v8f o0 = z, o1 = z, o2 = z, o3 = z;

    // One 32-key block from the staged LDS tiles (fully masked blocks are
    // no-ops: cur=-inf -> corr=1, exp()=0, rs=0, state unchanged).
    auto process = [&](const _Float16* cK, const _Float16* cV, int j0) {
        v8f s0 = z, s1 = z;
        {
            v16h ka = load_frag_a(cK, 0, 0,  BPITCH, lane);
            s0 = wmma_16x16x32(ka, bq_lo, s0);
            ka = load_frag_a(cK, 0, 32, BPITCH, lane);
            s0 = wmma_16x16x32(ka, bq_hi, s0);
            ka = load_frag_a(cK, 16, 0,  BPITCH, lane);
            s1 = wmma_16x16x32(ka, bq_lo, s1);
            ka = load_frag_a(cK, 16, 32, BPITCH, lane);
            s1 = wmma_16x16x32(ka, bq_hi, s1);
        }
        float cur = NEG_INF;
#pragma unroll
        for (int v = 0; v < 8; ++v) {
            const int key0 = j0 + hi8 + v;
            const int key1 = key0 + 16;
            float x0 = (key0 <= qcol) ? s0[v] : NEG_INF;
            float x1 = (key1 <= qcol) ? s1[v] : NEG_INF;
            s0[v] = x0; s1[v] = x1;
            cur = fmaxf(cur, fmaxf(x0, x1));
        }
        cur = fmaxf(cur, __shfl_xor(cur, 16, 32));
        const float mnew = fmaxf(m_i, cur);
        const float corr = __expf(m_i - mnew);
        float rs = 0.f;
#pragma unroll
        for (int v = 0; v < 8; ++v) {
            const float e0 = __expf(s0[v] - mnew);
            const float e1 = __expf(s1[v] - mnew);
            s0[v] = e0; s1[v] = e1;
            rs += e0 + e1;
        }
        rs += __shfl_xor(rs, 16, 32);
        l_i = l_i * corr + rs;
        m_i = mnew;
#pragma unroll
        for (int v = 0; v < 8; ++v) {
            o0[v] *= corr; o1[v] *= corr; o2[v] *= corr; o3[v] *= corr;
        }
        v16h pb;
#pragma unroll
        for (int v = 0; v < 8; ++v) {
            const float sh0 = __shfl_xor(s0[v], 16, 32);
            const float sh1 = __shfl_xor(s1[v], 16, 32);
            pb[v]     = (_Float16)(isHi ? sh1   : s0[v]);
            pb[8 + v] = (_Float16)(isHi ? s1[v] : sh0);
        }
        v16h va = load_frag_a(cV, 0,  0, VPITCH, lane);
        o0 = wmma_16x16x32(va, pb, o0);
        va = load_frag_a(cV, 16, 0, VPITCH, lane);
        o1 = wmma_16x16x32(va, pb, o1);
        va = load_frag_a(cV, 32, 0, VPITCH, lane);
        o2 = wmma_16x16x32(va, pb, o2);
        va = load_frag_a(cV, 48, 0, VPITCH, lane);
        o3 = wmma_16x16x32(va, pb, o3);
    };

    const int jend = blockIdx.x * 64 + 63;     // block-uniform causal bound
    flash_stage(Kb, Vb, 0, sK[0], sV[0], tid, wave);
    stage_wait();
    __syncthreads();
#pragma unroll 1
    for (int j0 = 0; j0 <= jend; j0 += 64) {
        flash_stage(Kb, Vb, j0 + 32, sK[1], sV[1], tid, wave);
        process(sK[0], sV[0], j0);
        stage_wait();
        __syncthreads();
        if (j0 + 64 <= jend)
            flash_stage(Kb, Vb, j0 + 64, sK[0], sV[0], tid, wave);
        process(sK[1], sV[1], j0 + 32);
        stage_wait();
        __syncthreads();
    }

    // --- epilogue: O[t][d] = O^T / l, packed b128 stores ---
    const float inv = 1.0f / l_i;
    const int b = bh / NHEAD, h = bh % NHEAD;
    const int t = q0 + (lane & 15);
    _Float16* row = AO + ((size_t)(b * SEQ_T + t)) * EMB_C + h * HSIZE;
    v8h r0, r1, r2, r3;
#pragma unroll
    for (int v = 0; v < 8; ++v) {
        r0[v] = (_Float16)(o0[v] * inv);
        r1[v] = (_Float16)(o1[v] * inv);
        r2[v] = (_Float16)(o2[v] * inv);
        r3[v] = (_Float16)(o3[v] * inv);
    }
    *(v8h*)(row +  0 + hi8) = r0;
    *(v8h*)(row + 16 + hi8) = r1;
    *(v8h*)(row + 32 + hi8) = r2;
    *(v8h*)(row + 48 + hi8) = r3;
}

// ----- Kernel 4: out = AO @ w_proj^T + b_proj (f32 output) -----------------

__global__ __launch_bounds__(256)
void proj_gemm(const _Float16* __restrict__ AO,
               const _Float16* __restrict__ wph,
               const float*    __restrict__ b_proj,
               float* __restrict__ out) {
    __shared__ __align__(16) _Float16 sB[2][64 * BPITCH];
    const int tid  = threadIdx.x;
    const int lane = tid & 31;
    const int wave = tid >> 5;
    const int m0 = blockIdx.x * 256 + wave * 32;
    const int n0 = blockIdx.y * 64;

    const v8f z = {0.f,0.f,0.f,0.f,0.f,0.f,0.f,0.f};
    v8f acc[2][4];
#pragma unroll
    for (int mh = 0; mh < 2; ++mh)
#pragma unroll
        for (int t = 0; t < 4; ++t) acc[mh][t] = z;

    gemm_mainloop(AO, wph, m0, n0, EMB_C, sB, acc, tid, lane, wave);

    const int hi8 = (lane >> 4) * 8;
    const int nc  = lane & 15;
#pragma unroll
    for (int mh = 0; mh < 2; ++mh) {
#pragma unroll
        for (int t = 0; t < 4; ++t) {
            const int d = n0 + t * 16 + nc;
            const float bias = b_proj[d];
#pragma unroll
            for (int v = 0; v < 8; ++v) {
                const int r = m0 + mh * 16 + hi8 + v;
                out[(size_t)r * EMB_C + d] = acc[mh][t][v] + bias;
            }
        }
    }
}

// ---------------------------------------------------------------------------

extern "C" void kernel_launch(void* const* d_in, const int* in_sizes, int n_in,
                              void* d_out, int out_size, void* d_ws, size_t ws_size,
                              hipStream_t stream) {
    (void)in_sizes; (void)n_in; (void)out_size; (void)ws_size;

    const float* x      = (const float*)d_in[0];
    const float* w_attn = (const float*)d_in[1];
    const float* b_attn = (const float*)d_in[2];
    const float* w_proj = (const float*)d_in[3];
    const float* b_proj = (const float*)d_in[4];
    float* out = (float*)d_out;

    char* ws = (char*)d_ws;
    const size_t n_x  = (size_t)MTOT * EMB_C;
    const size_t n_wa = (size_t)QKV_N * EMB_C;
    const size_t n_wp = (size_t)EMB_C * EMB_C;

    size_t off = 0;
    auto take = [&](size_t elems) {
        _Float16* p = (_Float16*)(ws + off);
        off += ((elems * sizeof(_Float16)) + 255) & ~(size_t)255;
        return p;
    };
    _Float16* xh  = take(n_x);
    _Float16* wah = take(n_wa);
    _Float16* wph = take(n_wp);
    _Float16* Qh  = take(n_x);
    _Float16* Kh  = take(n_x);
    _Float16* Vt  = take(n_x);
    _Float16* AO  = take(n_x);

    cvt_f32_to_f16<<<4096, 256, 0, stream>>>(x,      xh,  (int)n_x);
    cvt_f32_to_f16<<<2048, 256, 0, stream>>>(w_attn, wah, (int)n_wa);
    cvt_f32_to_f16<<<1024, 256, 0, stream>>>(w_proj, wph, (int)n_wp);

    qkv_gemm<<<dim3(MTOT / 256, QKV_N / 64), 256, 0, stream>>>(
        xh, wah, b_attn, Qh, Kh, Vt);

    flash_attn<<<dim3(SEQ_T / 64, BATCH * NHEAD), 128, 0, stream>>>(
        Qh, Kh, Vt, AO);

    proj_gemm<<<dim3(MTOT / 256, EMB_C / 64), 256, 0, stream>>>(
        AO, wph, b_proj, out);
}
